// GraphConvLayer_54245436948964
// MI455X (gfx1250) — matrix operand order repaired and checked by reading, not compiled, not run on hardware
//
#include <hip/hip_runtime.h>

typedef __attribute__((ext_vector_type(16))) __bf16   v16bf;
typedef __attribute__((ext_vector_type(8)))  unsigned v8u;
typedef __attribute__((ext_vector_type(4)))  unsigned v4u;
typedef __attribute__((ext_vector_type(8)))  float    v8f;

#define BB     32
#define CC     64
#define TT     64
#define VV     325
#define VPAD   352          // 11 K-chunks of 32
#define NTILES 21           // ceil(325/16)
#define NKC    11
#define NWAVES 8
#define TV     (TT * VV)
#define GMAX   (VV * VV)
#define NGFRAG (NTILES * NKC)     // 231 gso fragments
#define NFRAG  (NGFRAG + 16)      // + 16 weight fragments

// RNE pack (prep kernel only; runs once)
__device__ __forceinline__ unsigned pk_bf16(float f0, float f1) {
    unsigned u0 = __builtin_bit_cast(unsigned, f0);
    unsigned u1 = __builtin_bit_cast(unsigned, f1);
    u0 += 0x7FFFu + ((u0 >> 16) & 1u);
    u1 += 0x7FFFu + ((u1 >> 16) & 1u);
    return (u0 >> 16) | (u1 & 0xFFFF0000u);
}

// fast truncating pack: single v_perm_b32
__device__ __forceinline__ unsigned pack2(float f0, float f1) {
    return __builtin_amdgcn_perm(__builtin_bit_cast(unsigned, f1),
                                 __builtin_bit_cast(unsigned, f0), 0x07060302u);
}

__device__ __forceinline__ v8f wmma_bf16(v16bf a, v16bf b, v8f c) {
    return __builtin_amdgcn_wmma_f32_16x16x32_bf16(
        false, a, false, b, (short)0, c, false, false);
}

// Load the 8 weight fragments of one kc (8 consecutive 1KB blocks) in one
// asm block: 16 x global_load_b128 at immediate offsets, single wait.
// asm volatile => cannot be hoisted/spilled across tiles.
__device__ __forceinline__ void gload_w8(const unsigned* p, v16bf W[8]) {
    v4u a[8], b[8];
    asm volatile(
        "global_load_b128 %0, %16, off\n\t"
        "global_load_b128 %8, %16, off offset:16\n\t"
        "global_load_b128 %1, %16, off offset:1024\n\t"
        "global_load_b128 %9, %16, off offset:1040\n\t"
        "global_load_b128 %2, %16, off offset:2048\n\t"
        "global_load_b128 %10, %16, off offset:2064\n\t"
        "global_load_b128 %3, %16, off offset:3072\n\t"
        "global_load_b128 %11, %16, off offset:3088\n\t"
        "global_load_b128 %4, %16, off offset:4096\n\t"
        "global_load_b128 %12, %16, off offset:4112\n\t"
        "global_load_b128 %5, %16, off offset:5120\n\t"
        "global_load_b128 %13, %16, off offset:5136\n\t"
        "global_load_b128 %6, %16, off offset:6144\n\t"
        "global_load_b128 %14, %16, off offset:6160\n\t"
        "global_load_b128 %7, %16, off offset:7168\n\t"
        "global_load_b128 %15, %16, off offset:7184\n\t"
        "s_wait_loadcnt 0"
        : "=v"(a[0]), "=v"(a[1]), "=v"(a[2]), "=v"(a[3]),
          "=v"(a[4]), "=v"(a[5]), "=v"(a[6]), "=v"(a[7]),
          "=v"(b[0]), "=v"(b[1]), "=v"(b[2]), "=v"(b[3]),
          "=v"(b[4]), "=v"(b[5]), "=v"(b[6]), "=v"(b[7])
        : "v"(p) : "memory");
#pragma unroll
    for (int j = 0; j < 8; ++j)
        W[j] = __builtin_bit_cast(v16bf,
                   __builtin_shufflevector(a[j], b[j], 0, 1, 2, 3, 4, 5, 6, 7));
}

// ---------------------------------------------------------------------------
// Prep: gso + weight -> WMMA-fragment-ordered bf16 blobs in ws (253 KB).
// ---------------------------------------------------------------------------
__global__ __launch_bounds__(32) void prep_frags(
    const float* __restrict__ gso, const float* __restrict__ weight,
    unsigned* __restrict__ ws)
{
    const int lane = threadIdx.x & 31;
    const int lo = lane & 15, hi = lane >> 4;
    const int blk = blockIdx.x;
    v8u u;
    if (blk < NGFRAG) {
        const int mt = blk / NKC, kc = blk - mt * NKC;
        int h = mt * 16 + lo;  h = (h < VV) ? h : (VV - 1);
        const int grow = h * VV;
#pragma unroll
        for (int p = 0; p < 8; ++p) {
            int k  = kc * 32 + 2 * p + 16 * hi;
            int i0 = grow + k;     i0 = (i0 < GMAX) ? i0 : (GMAX - 1);
            int i1 = grow + k + 1; i1 = (i1 < GMAX) ? i1 : (GMAX - 1);
            u[p] = pk_bf16(gso[i0], gso[i1]);
        }
    } else {
        const int f  = blk - NGFRAG;                 // [kc][w][nt]
        const int kc = f >> 3, w = (f >> 2) & 1, nt = f & 3;
        const int n  = nt * 16 + lo;
#pragma unroll
        for (int p = 0; p < 8; ++p) {
            int base = w * (CC * CC) + (kc * 32 + 2 * p + 16 * hi) * CC + n;
            u[p] = pk_bf16(weight[base], weight[base + CC]);
        }
    }
    *(v8u*)(ws + (size_t)blk * 256 + lane * 8) = u;
}

// ---------------------------------------------------------------------------
// Main fused kernel: one workgroup (8 wave32) per (b,t).
// ---------------------------------------------------------------------------
__global__ __launch_bounds__(256) void graphconv_fused(
    const float* __restrict__ x,      // [B,C,T,V]
    const unsigned* __restrict__ ws,  // prepped fragments
    const float* __restrict__ bias,   // [C]
    float* __restrict__ out)          // [B,C,T,V]
{
    __shared__ unsigned short lds_xp[CC * VPAD];        // Xp bf16, [c][v]
    __shared__ unsigned short lds_st[NWAVES * 16 * CC]; // per-wave X1 tile, [c][vloc]

    const int tid  = threadIdx.x;
    const int lo   = tid & 15;
    const int hi   = (tid >> 4) & 1;
    const int lane = tid & 31;
    // wave id as SGPR: wave-uniform branches, scalar address math, full EXEC at WMMA
    const int wvu  = __builtin_amdgcn_readfirstlane(tid >> 5);

    const int bt  = blockIdx.x;
    const int b   = bt >> 6;
    const int t   = bt & (TT - 1);
    const int xbt = (b * CC) * TV + t * VV;

    // ---- Xp = x[b,:,t,:] -> LDS bf16 [c][v] ----
    {
        unsigned* xpw = (unsigned*)lds_xp;
        for (int i = tid; i < CC * (VPAD / 2); i += 256) {
            int c  = i / (VPAD / 2);
            int v2 = (i - c * (VPAD / 2)) * 2;
            float f0 = (v2     < VV) ? x[xbt + c * TV + v2]     : 0.0f;
            float f1 = (v2 + 1 < VV) ? x[xbt + c * TV + v2 + 1] : 0.0f;
            xpw[i] = pack2(f0, f1);
        }
    }
    __syncthreads();

    const unsigned* xp32  = (const unsigned*)lds_xp;
    const v8u*      frags = (const v8u*)ws;

    float bsv[4];
#pragma unroll
    for (int nt = 0; nt < 4; ++nt) bsv[nt] = bias[nt * 16 + lo];

    const int  mt0 = wvu, mt1 = wvu + 8, mt2 = wvu + 16;
    const bool has3 = (mt2 < NTILES);                 // scalar condition

    // prefetch residual/output cachelines (the real HBM traffic)
    {
        const float* px = x + ((b * CC + lo) * TT + t) * VV + 8 * hi;
#pragma unroll
        for (int nt = 0; nt < 4; ++nt) {
            __builtin_prefetch(px + nt * 16 * TV + mt0 * 16, 0, 1);
            __builtin_prefetch(px + nt * 16 * TV + mt1 * 16, 0, 1);
            if (has3) __builtin_prefetch(px + nt * 16 * TV + mt2 * 16, 0, 1);
        }
    }

    auto xpB = [&](int kc, int nt) -> v16bf {   // pairs contiguous -> ds_load_b128
        v8u u;
        int base = ((nt * 16 + lo) * VPAD + kc * 32 + 16 * hi) >> 1;
#pragma unroll
        for (int p = 0; p < 8; ++p) u[p] = xp32[base + p];
        return __builtin_bit_cast(v16bf, u);
    };
    auto ldA = [&](int mt, int kc) -> v16bf {   // SGPR base + lane offset
        return __builtin_bit_cast(v16bf, frags[(mt * NKC + kc) * 32 + lane]);
    };

    // ===== stage 1: kc-outer, B-fragments shared across my 2-3 row tiles =====
    v8f xacc[3][4] = {};
    for (int kc = 0; kc < NKC; ++kc) {
        v16bf Bf[4];
#pragma unroll
        for (int nt = 0; nt < 4; ++nt) Bf[nt] = xpB(kc, nt);
        v16bf A0 = ldA(mt0, kc);
        v16bf A1 = ldA(mt1, kc);
#pragma unroll
        for (int nt = 0; nt < 4; ++nt) {
            xacc[0][nt] = wmma_bf16(A0, Bf[nt], xacc[0][nt]);
            xacc[1][nt] = wmma_bf16(A1, Bf[nt], xacc[1][nt]);
        }
        if (has3) {                               // s_cbranch (uniform)
            v16bf A2 = ldA(mt2, kc);
#pragma unroll
            for (int nt = 0; nt < 4; ++nt)
                xacc[2][nt] = wmma_bf16(A2, Bf[nt], xacc[2][nt]);
        }
    }

    // ===== per-tile: stage X1, stage-2 matmul, epilogue =====
    auto process = [&](const v8f (&xa)[4], int mt) {
        // staging layout [c][vloc]: D-pairs (r,r+1) lane-local -> 16x v_perm + ds_store_b32
        const int stb   = wvu * (16 * CC);        // u16 units
        unsigned* stw   = (unsigned*)lds_st;
        const int stb32 = wvu * (8 * CC);         // u32 units
#pragma unroll
        for (int nt = 0; nt < 4; ++nt) {
            int c8 = (nt * 16 + lo) * 8 + 4 * hi;
#pragma unroll
            for (int r = 0; r < 8; r += 2)
                stw[stb32 + c8 + r / 2] = pack2(xa[nt][r], xa[nt][r + 1]);
        }

        v8f oacc[4] = {};
        const int vrow = mt * 16 + lo;
#pragma unroll
        for (int kc = 0; kc < 2; ++kc) {
            v8u a0u, a1u;
#pragma unroll
            for (int p = 0; p < 8; ++p) {
                int k = kc * 32 + 2 * p + 16 * hi;
                a0u[p] = (unsigned)lds_xp[k * VPAD + vrow]            // Xp rows
                       | ((unsigned)lds_xp[(k + 1) * VPAD + vrow] << 16);
                a1u[p] = (unsigned)lds_st[stb + k * 16 + lo]          // staged X1
                       | ((unsigned)lds_st[stb + (k + 1) * 16 + lo] << 16);
            }
            v16bf A0 = __builtin_bit_cast(v16bf, a0u);
            v16bf A1 = __builtin_bit_cast(v16bf, a1u);

            v16bf W[8];                           // one asm block: 16 loads, 1 wait
            gload_w8((const unsigned*)&frags[(NGFRAG + kc * 8) * 32 + lane], W);
#pragma unroll
            for (int nt = 0; nt < 4; ++nt) {
                oacc[nt] = wmma_bf16(A0, W[nt],     oacc[nt]);
                oacc[nt] = wmma_bf16(A1, W[4 + nt], oacc[nt]);
            }
        }

        if (mt < NTILES - 1) {                    // full tile, no predication
#pragma unroll
            for (int nt = 0; nt < 4; ++nt) {
                int obase = ((b * CC + nt * 16 + lo) * TT + t) * VV + mt * 16 + 8 * hi;
#pragma unroll
                for (int r = 0; r < 8; ++r)
                    out[obase + r] = oacc[nt][r] + bsv[nt] + x[obase + r];
            }
        } else {                                  // single partial tile
#pragma unroll
            for (int nt = 0; nt < 4; ++nt) {
                int obase = ((b * CC + nt * 16 + lo) * TT + t) * VV;
#pragma unroll
                for (int r = 0; r < 8; ++r) {
                    int v = mt * 16 + r + 8 * hi;
                    if (v < VV)
                        out[obase + v] = oacc[nt][r] + bsv[nt] + x[obase + v];
                }
            }
        }
    };

    process(xacc[0], mt0);
    process(xacc[1], mt1);
    if (has3) process(xacc[2], mt2);
}

extern "C" void kernel_launch(void* const* d_in, const int* in_sizes, int n_in,
                              void* d_out, int out_size, void* d_ws, size_t ws_size,
                              hipStream_t stream) {
    const float* x      = (const float*)d_in[0];
    const float* gso    = (const float*)d_in[1];
    const float* weight = (const float*)d_in[2];
    const float* bias   = (const float*)d_in[3];
    float*       out    = (float*)d_out;
    unsigned*    ws     = (unsigned*)d_ws;    // needs NFRAG KiB = 253 KB

    hipLaunchKernelGGL(prep_frags, dim3(NFRAG), dim3(32), 0, stream,
                       gso, weight, ws);
    hipLaunchKernelGGL(graphconv_fused, dim3(BB * TT), dim3(256), 0, stream,
                       x, ws, bias, out);
}
